// GNN_83373905150176
// MI455X (gfx1250) — matrix operand order repaired and checked by reading, not compiled, not run on hardware
//
#include <hip/hip_runtime.h>
#include <hip/hip_bf16.h>
#include <math.h>

// ---------------- problem constants ----------------
#define BSZ   32
#define NPB   200
#define NN    (BSZ * NPB)     // 6400 nodes
#define EE    204800          // edges
#define CP    1024
#define HID   512
#define HIDH  (HID / 2)       // 256
#define NTY   4
#define EDIM  46              // edge_attr dim
#define EDP   64              // padded K for edge MLP stage 1
#define KX    (CP + HIDH)     // 1280
#define ETILE 64              // edges per block in fused edge MLP

typedef _Float16 h8   __attribute__((ext_vector_type(8)));
typedef _Float16 v16h __attribute__((ext_vector_type(16)));
typedef float    v8f  __attribute__((ext_vector_type(8)));

// ---------------- WMMA fragment loaders (wave32) ----------------
// A (16x32 f16, row-major source, leading dim ld):
//   lanes 0-15: M=lane,   K = k0+0..7 (elems 0-7), k0+16..23 (elems 8-15)
//   lanes16-31: M=lane-16,K = k0+8..15,            k0+24..31
__device__ __forceinline__ v16h load_fragA(const _Float16* base, int ld, int row0, int k0) {
    int lane = threadIdx.x & 31;
    int r    = lane & 15;
    int half = lane >> 4;
    const _Float16* p = base + (size_t)(row0 + r) * ld + k0 + half * 8;
    h8 lo = *(const h8*)p;
    h8 hi = *(const h8*)(p + 16);
    return __builtin_shufflevector(lo, hi, 0,1,2,3,4,5,6,7,8,9,10,11,12,13,14,15);
}

// B (32x16 f16) loaded from pre-transposed weights Bt (N x K row-major):
//   lanes 0-15: N=lane,   K = k0+0..7 (elems 0-7), k0+16..23 (elems 8-15)
//   lanes16-31: N=lane-16,K = k0+8..15,            k0+24..31
__device__ __forceinline__ v16h load_fragB(const _Float16* bt, int ldk, int col0, int k0) {
    int lane = threadIdx.x & 31;
    int n    = lane & 15;
    int half = lane >> 4;
    const _Float16* p = bt + (size_t)(col0 + n) * ldk + k0 + half * 8;
    h8 lo = *(const h8*)p;
    h8 hi = *(const h8*)(p + 16);
    return __builtin_shufflevector(lo, hi, 0,1,2,3,4,5,6,7,8,9,10,11,12,13,14,15);
}

// float atomic max via sign-split int atomics (init must be -inf)
__device__ __forceinline__ void atomicMaxF32(float* addr, float val) {
    if (val >= 0.0f) atomicMax((int*)addr, __float_as_int(val));
    else             atomicMin((unsigned int*)addr, __float_as_uint(val));
}

// ---------------- workspace init ----------------
__global__ __launch_bounds__(256) void k_init(float* amax, float* denom, float* accum) {
    size_t i = (size_t)blockIdx.x * 256 + threadIdx.x;
    if (i < (size_t)NN * HID) accum[i] = 0.0f;
    if (i < NN) { amax[i] = -INFINITY; denom[i] = 0.0f; }
}

// ---------------- weight transpose + f16 convert (W: K x Nc -> Wt: Nc x KP) ----------------
__global__ __launch_bounds__(256) void k_transpose(const float* W, _Float16* Wt,
                                                   int K, int Nc, int KP) {
    int idx = blockIdx.x * 256 + threadIdx.x;
    if (idx >= Nc * KP) return;
    int n = idx / KP, k = idx - n * KP;
    Wt[idx] = (k < K) ? (_Float16)W[(size_t)k * Nc + n] : (_Float16)0.0f;
}

// v = W_edge @ att_edge  (512-vector)
__global__ __launch_bounds__(256) void k_vedge(const float* W_edge, const float* att_edge, float* v) {
    int i = blockIdx.x * 256 + threadIdx.x;
    if (i >= HID) return;
    float s = 0.0f;
    for (int j = 0; j < HID; ++j) s += W_edge[(size_t)i * HID + j] * att_edge[j];
    v[i] = s;
}

// ---------------- build node input matrix A (N x 1280, f16, post-relu) ----------------
__global__ __launch_bounds__(256) void k_prepA(const float* x, const float* qa,
                                               const float* ntypes, const float* nscore,
                                               const float* W_ntype, const float* b_ntype,
                                               _Float16* A) {
    int i = blockIdx.x;                 // node
    int tid = threadIdx.x;
    const float* xs = ((i % NPB) == 0) ? &qa[(size_t)(i / NPB) * CP] : &x[(size_t)i * CP];
    for (int c = tid; c < CP; c += 256)
        A[(size_t)i * KX + c] = (_Float16)fmaxf(xs[c], 0.0f);
    float t0 = ntypes[i * NTY + 0], t1 = ntypes[i * NTY + 1];
    float t2 = ntypes[i * NTY + 2], t3 = ntypes[i * NTY + 3];
    float sc = nscore[i];
    for (int c = tid; c < HIDH; c += 256) {
        float v = b_ntype[c]
                + t0 * W_ntype[0 * HIDH + c] + t1 * W_ntype[1 * HIDH + c]
                + t2 * W_ntype[2 * HIDH + c] + t3 * W_ntype[3 * HIDH + c]
                + sc * W_ntype[4 * HIDH + c];
        A[(size_t)i * KX + CP + c] = (_Float16)fmaxf(v, 0.0f);
    }
}

// ---------------- node GEMM: C(M x 512) = [relu](A @ Bt^T + bias) ----------------
// RT row-tiles per block (B fragments held in registers, reused across row tiles).
// grid.x = M/(16*RT); 8 waves/block, wave w owns 64 output cols.
template <int KDIM, int RT, bool RELU, bool F16OUT>
__global__ __launch_bounds__(256) void k_gemm(const _Float16* __restrict__ A,
                                              const _Float16* __restrict__ Bt,
                                              const float* __restrict__ bias,
                                              void* __restrict__ Cout) {
    int row0 = blockIdx.x * (16 * RT);
    int w    = threadIdx.x >> 5;
    int col0 = w * 64;
    v8f acc[RT][4] = {};
    for (int k = 0; k < KDIM; k += 32) {
        v16h b[4];
#pragma unroll
        for (int j = 0; j < 4; ++j) b[j] = load_fragB(Bt, KDIM, col0 + 16 * j, k);
#pragma unroll
        for (int rt = 0; rt < RT; ++rt) {
            v16h a = load_fragA(A, KDIM, row0 + 16 * rt, k);
#pragma unroll
            for (int j = 0; j < 4; ++j)
                acc[rt][j] = __builtin_amdgcn_wmma_f32_16x16x32_f16(false, a, false, b[j],
                                                                    (short)0, acc[rt][j],
                                                                    false, false);
        }
    }
    int lane = threadIdx.x & 31;
    int c    = lane & 15;
    int half = lane >> 4;
#pragma unroll
    for (int rt = 0; rt < RT; ++rt)
#pragma unroll
        for (int j = 0; j < 4; ++j) {
            int col  = col0 + 16 * j + c;
            float bv = bias ? bias[col] : 0.0f;
#pragma unroll
            for (int r = 0; r < 8; ++r) {
                float v = acc[rt][j][r] + bv;
                if (RELU) v = fmaxf(v, 0.0f);
                size_t o = (size_t)(row0 + 16 * rt + r + 8 * half) * HID + col;
                if (F16OUT) ((_Float16*)Cout)[o] = (_Float16)v;
                else        ((float*)Cout)[o]    = v;
            }
        }
}

// ---------------- a_src / a_dst dots (1 wave per node) ----------------
__global__ __launch_bounds__(256) void k_dots(const float* hl, const float* att_src,
                                              const float* att_dst, float* a_src, float* a_dst) {
    int node = (blockIdx.x * 256 + threadIdx.x) >> 5;
    int lane = threadIdx.x & 31;
    if (node >= NN) return;
    float s1 = 0.0f, s2 = 0.0f;
    const float* h = &hl[(size_t)node * HID];
    for (int c = lane; c < HID; c += 32) { float v = h[c]; s1 += v * att_src[c]; s2 += v * att_dst[c]; }
    for (int s = 16; s > 0; s >>= 1) { s1 += __shfl_down(s1, s, 32); s2 += __shfl_down(s2, s, 32); }
    if (lane == 0) { a_src[node] = s1; a_dst[node] = s2; }
}

// ---------------- fused edge MLP: a_edge = relu(relu(attr@We1+b1)@We2+b2) . v ----------------
// 1 block (8 waves) per 64 edges; ea1 lives in LDS (64x512 f16 = 64KB, never hits HBM).
// B fragments held in registers across the 4 row tiles -> 4x less L2 weight traffic.
__global__ __launch_bounds__(256) void k_edge_mlp(const float* __restrict__ edge_attr,
                                                  const _Float16* __restrict__ We1T,
                                                  const float* __restrict__ be1,
                                                  const _Float16* __restrict__ We2T,
                                                  const float* __restrict__ be2,
                                                  const float* __restrict__ vedge,
                                                  float* __restrict__ a_edge) {
    __shared__ __align__(16) _Float16 s_attr[ETILE * EDP];   // 8 KB
    __shared__ __align__(16) _Float16 s_ea1[ETILE * HID];    // 64 KB
    __shared__ float s_ae[ETILE];
    int tid = threadIdx.x;
    int e0  = blockIdx.x * ETILE;

    // stage 0: load 64 x 46 attr tile -> f16, zero-pad to K=64
    for (int i = tid; i < ETILE * EDP; i += 256) {
        int r = i >> 6, c = i & 63;
        float v = (c < EDIM) ? edge_attr[(size_t)(e0 + r) * EDIM + c] : 0.0f;
        s_attr[i] = (_Float16)v;
    }
    if (tid < ETILE) {
        s_ae[tid] = 0.0f;
        if (blockIdx.x + 1 < gridDim.x)
            __builtin_prefetch(&edge_attr[(size_t)(e0 + ETILE + tid) * EDIM], 0, 0);
    }
    __syncthreads();

    int w = tid >> 5, lane = tid & 31, c = lane & 15, half = lane >> 4;
    int col0 = w * 64;

    // stage 1: ea1 = relu(attr @ We1 + be1), 64 x 512 into LDS
    {
        v8f acc[4][4] = {};
        for (int k = 0; k < EDP; k += 32) {
            v16h b[4];
#pragma unroll
            for (int j = 0; j < 4; ++j) b[j] = load_fragB(We1T, EDP, col0 + 16 * j, k);
#pragma unroll
            for (int rt = 0; rt < 4; ++rt) {
                v16h a = load_fragA(s_attr, EDP, 16 * rt, k);
#pragma unroll
                for (int j = 0; j < 4; ++j)
                    acc[rt][j] = __builtin_amdgcn_wmma_f32_16x16x32_f16(false, a, false, b[j],
                                                                        (short)0, acc[rt][j],
                                                                        false, false);
            }
        }
#pragma unroll
        for (int rt = 0; rt < 4; ++rt)
#pragma unroll
            for (int j = 0; j < 4; ++j) {
                int col = col0 + 16 * j + c;
                float bv = be1[col];
#pragma unroll
                for (int r = 0; r < 8; ++r)
                    s_ea1[(16 * rt + r + 8 * half) * HID + col] =
                        (_Float16)fmaxf(acc[rt][j][r] + bv, 0.0f);
            }
    }
    __syncthreads();

    // stage 2: ea2 = relu(ea1 @ We2 + be2); immediately dot with v
    {
        v8f acc[4][4] = {};
        for (int k = 0; k < HID; k += 32) {
            v16h b[4];
#pragma unroll
            for (int j = 0; j < 4; ++j) b[j] = load_fragB(We2T, HID, col0 + 16 * j, k);
#pragma unroll
            for (int rt = 0; rt < 4; ++rt) {
                v16h a = load_fragA(s_ea1, HID, 16 * rt, k);
#pragma unroll
                for (int j = 0; j < 4; ++j)
                    acc[rt][j] = __builtin_amdgcn_wmma_f32_16x16x32_f16(false, a, false, b[j],
                                                                        (short)0, acc[rt][j],
                                                                        false, false);
            }
        }
#pragma unroll
        for (int rt = 0; rt < 4; ++rt) {
            float part[8] = {0.f, 0.f, 0.f, 0.f, 0.f, 0.f, 0.f, 0.f};
#pragma unroll
            for (int j = 0; j < 4; ++j) {
                int col = col0 + 16 * j + c;
                float bv = be2[col], vv = vedge[col];
#pragma unroll
                for (int r = 0; r < 8; ++r)
                    part[r] += fmaxf(acc[rt][j][r] + bv, 0.0f) * vv;
            }
            // reduce 16 column-lanes (xor stays inside each 16-lane half)
            for (int s = 1; s < 16; s <<= 1)
#pragma unroll
                for (int r = 0; r < 8; ++r) part[r] += __shfl_xor(part[r], s, 32);
            if (c == 0)
#pragma unroll
                for (int r = 0; r < 8; ++r)
                    atomicAdd(&s_ae[16 * rt + r + 8 * half], part[r]);
        }
    }
    __syncthreads();
    if (tid < ETILE) a_edge[e0 + tid] = s_ae[tid];
}

// ---------------- alpha + segment max ----------------
__global__ __launch_bounds__(256) void k_alpha(const int* src, const int* dst,
                                               const float* a_src, const float* a_dst,
                                               const float* a_edge, float* alpha, float* amax) {
    int e = blockIdx.x * 256 + threadIdx.x;
    if (e >= EE) return;
    float al = a_src[src[e]] + a_dst[dst[e]] + a_edge[e];
    al = (al > 0.0f) ? al : 0.2f * al;          // leaky_relu(0.2)
    alpha[e] = al;
    atomicMaxF32(&amax[dst[e]], al);
}

// ---------------- exp + segment sum ----------------
__global__ __launch_bounds__(256) void k_exp(const int* dst, const float* alpha,
                                             const float* amax, float* exv, float* denom) {
    int e = blockIdx.x * 256 + threadIdx.x;
    if (e >= EE) return;
    int d = dst[e];
    float ex = __expf(alpha[e] - amax[d]);
    exv[e] = ex;
    atomicAdd(&denom[d], ex);
}

// ---------------- weighted scatter: accum[dst] += hl[src] * w (1 wave / edge) ----------------
__global__ __launch_bounds__(256) void k_scatter(const int* src, const int* dst,
                                                 const float* exv, const float* denom,
                                                 const float* hl, float* accum) {
    int e    = (blockIdx.x * 256 + threadIdx.x) >> 5;
    int lane = threadIdx.x & 31;
    if (e >= EE) return;
    int s = src[e], d = dst[e];
    float wgt = exv[e] / (denom[d] + 1e-16f);
    const float* hs = &hl[(size_t)s * HID];
    float*       ad = &accum[(size_t)d * HID];
    for (int c = lane; c < HID; c += 32) atomicAdd(&ad[c], hs[c] * wgt);
}

// ---------------- finalize: h0 rows + relu(mean pool), both +b_gat ----------------
__global__ __launch_bounds__(256) void k_final(const float* accum, const float* b_gat, float* out) {
    int g = blockIdx.x;
    for (int c = threadIdx.x; c < HID; c += 256) {
        float bb  = b_gat[c];
        float h0  = accum[(size_t)g * NPB * HID + c] + bb;
        float sum = 0.0f;
        for (int i = 0; i < NPB; ++i) sum += accum[(size_t)(g * NPB + i) * HID + c];
        out[g * HID + c]             = h0;
        out[BSZ * HID + g * HID + c] = fmaxf(sum * (1.0f / NPB) + bb, 0.0f);
    }
}

// ---------------- host orchestration ----------------
extern "C" void kernel_launch(void* const* d_in, const int* in_sizes, int n_in,
                              void* d_out, int out_size, void* d_ws, size_t ws_size,
                              hipStream_t stream) {
    const float* qa_emb    = (const float*)d_in[0];
    const float* x         = (const float*)d_in[1];
    const float* ntypes    = (const float*)d_in[3];
    const float* nscore    = (const float*)d_in[4];
    const int*   edge_idx  = (const int*)  d_in[5];
    const float* edge_attr = (const float*)d_in[7];
    const float* W_ntype   = (const float*)d_in[9];
    const float* b_ntype   = (const float*)d_in[10];
    const float* W_x2h     = (const float*)d_in[11];
    const float* b_x2h     = (const float*)d_in[12];
    const float* We1       = (const float*)d_in[13];
    const float* be1       = (const float*)d_in[14];
    const float* We2       = (const float*)d_in[15];
    const float* be2       = (const float*)d_in[16];
    const float* W_gat     = (const float*)d_in[17];
    const float* att_src   = (const float*)d_in[18];
    const float* att_dst   = (const float*)d_in[19];
    const float* W_edge    = (const float*)d_in[20];
    const float* att_edge  = (const float*)d_in[21];
    const float* b_gat     = (const float*)d_in[22];
    const int* src = edge_idx;
    const int* dst = edge_idx + EE;
    float* out = (float*)d_out;

    // carve workspace (256B aligned)
    char* p = (char*)d_ws;
    auto carve = [&](size_t bytes) { char* r = p; p += (bytes + 255) & ~(size_t)255; return r; };
    _Float16* A_node = (_Float16*)carve((size_t)NN * KX * 2);
    _Float16* Wx2hT  = (_Float16*)carve((size_t)HID * KX * 2);
    _Float16* WgatT  = (_Float16*)carve((size_t)HID * HID * 2);
    _Float16* We1T   = (_Float16*)carve((size_t)HID * EDP * 2);
    _Float16* We2T   = (_Float16*)carve((size_t)HID * HID * 2);
    float*    vedge  = (float*)   carve(HID * 4);
    _Float16* h_f16  = (_Float16*)carve((size_t)NN * HID * 2);
    float*    hl     = (float*)   carve((size_t)NN * HID * 4);
    float*    a_src  = (float*)   carve(NN * 4);
    float*    a_dst  = (float*)   carve(NN * 4);
    float*    a_edge = (float*)   carve(EE * 4);
    float*    alpha  = (float*)   carve(EE * 4);
    float*    exv    = (float*)   carve(EE * 4);
    float*    amax   = (float*)   carve(NN * 4);
    float*    denom  = (float*)   carve(NN * 4);
    float*    accum  = (float*)   carve((size_t)NN * HID * 4);

    // 0) zero accumulators, init segment max to -inf
    k_init<<<(NN * HID + 255) / 256, 256, 0, stream>>>(amax, denom, accum);
    // 1) pre-transpose + f16-convert weights
    k_transpose<<<(HID * KX  + 255) / 256, 256, 0, stream>>>(W_x2h, Wx2hT, KX,  HID, KX);
    k_transpose<<<(HID * HID + 255) / 256, 256, 0, stream>>>(W_gat, WgatT, HID, HID, HID);
    k_transpose<<<(HID * EDP + 255) / 256, 256, 0, stream>>>(We1,   We1T,  EDIM, HID, EDP);
    k_transpose<<<(HID * HID + 255) / 256, 256, 0, stream>>>(We2,   We2T,  HID, HID, HID);
    k_vedge<<<(HID + 255) / 256, 256, 0, stream>>>(W_edge, att_edge, vedge);
    // 2) build node input matrix (qa substitution + x_extra + relu, f16)
    k_prepA<<<NN, 256, 0, stream>>>(x, qa_emb, ntypes, nscore, W_ntype, b_ntype, A_node);
    // 3) h = relu(A @ W_x2h + b)   [WMMA, 32-row tiles]
    k_gemm<KX, 2, true, true><<<NN / 32, 256, 0, stream>>>(A_node, Wx2hT, b_x2h, h_f16);
    // 4) hl = h @ W_gat            [WMMA, 32-row tiles]
    k_gemm<HID, 2, false, false><<<NN / 32, 256, 0, stream>>>(h_f16, WgatT, nullptr, hl);
    // 5) attention logits per node
    k_dots<<<NN / 8, 256, 0, stream>>>(hl, att_src, att_dst, a_src, a_dst);
    // 6) fused edge MLP -> a_edge  [WMMA, 64-edge tiles, LDS-resident ea1]
    k_edge_mlp<<<EE / ETILE, 256, 0, stream>>>(edge_attr, We1T, be1, We2T, be2, vedge, a_edge);
    // 7..9) segment softmax + weighted scatter
    k_alpha<<<EE / 256, 256, 0, stream>>>(src, dst, a_src, a_dst, a_edge, alpha, amax);
    k_exp<<<EE / 256, 256, 0, stream>>>(dst, alpha, amax, exv, denom);
    k_scatter<<<EE / 8, 256, 0, stream>>>(src, dst, exv, denom, hl, accum);
    // 10) outputs: h0 and pooled p
    k_final<<<BSZ, 256, 0, stream>>>(accum, b_gat, out);
    (void)in_sizes; (void)n_in; (void)out_size; (void)ws_size;
}
